// Attention_73289321939579
// MI455X (gfx1250) — matrix operand order repaired and checked by reading, not compiled, not run on hardware
//
#include <hip/hip_runtime.h>

typedef __attribute__((ext_vector_type(16))) __bf16 v16bf;
typedef __attribute__((ext_vector_type(8)))  __bf16 v8bf;
typedef __attribute__((ext_vector_type(8)))  float  v8f;

#define NB  2
#define NQ  2048
#define NC  512
#define NH  8
#define ND  64

static __device__ __forceinline__ v8f wmma_bf16(v16bf a, v16bf b, v8f c) {
  return __builtin_amdgcn_wmma_f32_16x16x32_bf16(false, a, false, b, (short)0, c,
                                                 false, false);
}

// A-fragment (16x32 bf16): two adjacent 16B runs (one contiguous 32B span)
static __device__ __forceinline__ v16bf ldA(const __bf16* p) {
  v8bf a0 = *(const v8bf*)(p);
  v8bf a1 = *(const v8bf*)(p + 8);
  v16bf r;
#pragma unroll
  for (int i = 0; i < 8; i++) { r[i] = a0[i]; r[8 + i] = a1[i]; }
  return r;
}
// B-fragment (32x16 bf16): one contiguous 32B run
static __device__ __forceinline__ v16bf ldB(const __bf16* p) {
  return *(const v16bf*)(p);
}

// elementwise fp32 -> bf16, 8 per thread (all sizes are multiples of 8)
__global__ void cvt_f32_bf16(const float* __restrict__ src,
                             __bf16* __restrict__ dst, int n) {
  int i = (blockIdx.x * blockDim.x + threadIdx.x) * 8;
  if (i < n) {
    float4 va = *(const float4*)(src + i);
    float4 vb = *(const float4*)(src + i + 4);
    v8bf o;
    o[0] = (__bf16)va.x; o[1] = (__bf16)va.y; o[2] = (__bf16)va.z; o[3] = (__bf16)va.w;
    o[4] = (__bf16)vb.x; o[5] = (__bf16)vb.y; o[6] = (__bf16)vb.z; o[7] = (__bf16)vb.w;
    *(v8bf*)(dst + i) = o;
  }
}

// Fused QKV + gate projection.
// A = qxB [4096, 512] bf16 row-major, B = WB [2048, 512] bf16 row-major
// (rows 0..1535 = w_qkv, rows 1536..2047 = w_g).
__global__ __launch_bounds__(128) void proj_qkvg(
    const __bf16* __restrict__ qxB, const __bf16* __restrict__ WB,
    const float* __restrict__ b_g,
    __bf16* __restrict__ qB, __bf16* __restrict__ kB,
    __bf16* __restrict__ vT, float* __restrict__ g) {
  const int waveId = blockIdx.x * 4 + (threadIdx.x >> 5);
  const int lane = threadIdx.x & 31;
  const int nn = lane & 15, hi = lane >> 4;
  const int mt = (waveId >> 5) << 4;  // 256 M-tiles of 16
  const int et = (waveId & 31) << 6;  // 32 E-groups of 64
  const int kbA = 16 * hi, kbB = 16 * hi;

  const __bf16* arow = qxB + (size_t)(mt + nn) * NC + kbA;
  auto loadFr = [&](int c0, v16bf& af, v16bf* bfv) {
    af = ldA(arow + c0);
#pragma unroll
    for (int t = 0; t < 4; t++)
      bfv[t] = ldB(WB + (size_t)(et + 16 * t + nn) * NC + c0 + kbB);
  };

  v8f acc[4] = {v8f{}, v8f{}, v8f{}, v8f{}};
  v16bf afA, afB, bfA[4], bfB[4];
  loadFr(0, afA, bfA);
  for (int c0 = 0; c0 < NC; c0 += 64) {
    loadFr(c0 + 32, afB, bfB);  // always in range (c0+32 <= 480)
#pragma unroll
    for (int t = 0; t < 4; t++) acc[t] = wmma_bf16(afA, bfA[t], acc[t]);
    loadFr((c0 + 64 < NC) ? c0 + 64 : 0, afA, bfA);  // dead after last iter
#pragma unroll
    for (int t = 0; t < 4; t++) acc[t] = wmma_bf16(afB, bfB[t], acc[t]);
  }

#pragma unroll
  for (int t = 0; t < 4; t++) {
    const int e = et + 16 * t + nn;
    const int region = (et + 16 * t) >> 9;  // uniform per tile (512 % 16 == 0)
#pragma unroll
    for (int r = 0; r < 8; r++) {
      const int m = mt + r + 8 * hi;
      const int b = m >> 11;
      const int q = m & (NQ - 1);
      float val = acc[t][r];
      if (region == 0) {
        int h = e >> 6, d = e & 63;
        qB[(((size_t)(b * NH + h)) * NQ + q) * ND + d] = (__bf16)(val * 0.125f);
      } else if (region == 1) {
        int e2 = e - 512, h = e2 >> 6, d = e2 & 63;
        kB[(((size_t)(b * NH + h)) * NQ + q) * ND + d] = (__bf16)val;
      } else if (region == 2) {
        int e2 = e - 1024, h = e2 >> 6, d = e2 & 63;
        vT[(((size_t)(b * NH + h)) * ND + d) * NQ + q] = (__bf16)val;
      } else {
        int e2 = e - 1536, h = e2 >> 6, d = e2 & 63;
        float s = val + b_g[e2];
        g[(((size_t)(b * NH + h)) * NQ + q) * ND + d] = 1.0f / (1.0f + __expf(-s));
      }
    }
  }
}

// element-wise max/sum reduction across 16-lane halves (rows of C tiles)
static __device__ __forceinline__ void redmax16(v8f& v) {
#pragma unroll
  for (int mask = 1; mask < 16; mask <<= 1)
#pragma unroll
    for (int r = 0; r < 8; r++) v[r] = fmaxf(v[r], __shfl_xor(v[r], mask, 16));
}
static __device__ __forceinline__ void redsum16(v8f& v) {
#pragma unroll
  for (int mask = 1; mask < 16; mask <<= 1)
#pragma unroll
    for (int r = 0; r < 8; r++) v[r] += __shfl_xor(v[r], mask, 16);
}

// Fused flash attention + bias + sigmoid gating.
// Grid: B*H*(Q/64) = 512 blocks; 4 waves/block; wave owns 16 queries.
__global__ __launch_bounds__(128) void attn_fused(
    const __bf16* __restrict__ qB, const __bf16* __restrict__ kB,
    const __bf16* __restrict__ vT, const float* __restrict__ g,
    const float* __restrict__ bias, __bf16* __restrict__ oG) {
  __shared__ __attribute__((aligned(16))) __bf16 plds[4 * 16 * 40];
  const int wave = threadIdx.x >> 5;
  const int lane = threadIdx.x & 31;
  const int nn = lane & 15, hi = lane >> 4;
  const int qblk = blockIdx.x & 31;
  const int bh = blockIdx.x >> 5;
  const int b = bh >> 3, h = bh & 7;
  const int qt = qblk * 64 + wave * 16;

  const __bf16* qbase = qB + (size_t)bh * NQ * ND;
  const __bf16* kbase = kB + (size_t)bh * NQ * ND;
  const __bf16* vbase = vT + (size_t)bh * ND * NQ;
  const float* gbase = g + (size_t)bh * NQ * ND;
  const float* bbase = bias + (size_t)b * NQ * NQ;
  __bf16* myl = plds + wave * (16 * 40);

  // Q A-fragments (d = 0..31, 32..63)
  v16bf qf[2];
  {
    const __bf16* qrow = qbase + (size_t)(qt + nn) * ND + 16 * hi;
    qf[0] = ldA(qrow);
    qf[1] = ldA(qrow + 32);
  }

  auto loadK = [&](int kt_, v16bf* kf) {
    const __bf16* kr0 = kbase + (size_t)(kt_ + nn) * ND + 16 * hi;
    const __bf16* kr1 = kbase + (size_t)(kt_ + 16 + nn) * ND + 16 * hi;
    kf[0] = ldB(kr0);
    kf[1] = ldB(kr0 + 32);
    kf[2] = ldB(kr1);
    kf[3] = ldB(kr1 + 32);
  };

  v8f O[4] = {v8f{}, v8f{}, v8f{}, v8f{}};
  v8f m, l = {};
#pragma unroll
  for (int r = 0; r < 8; r++) m[r] = -1e30f;

  v16bf kf[4];
  loadK(0, kf);

  for (int kt = 0; kt < NQ; kt += 32) {
    // issue V-fragment + bias loads early: they overlap the S WMMAs below
    v16bf vf[4];
#pragma unroll
    for (int t = 0; t < 4; t++)
      vf[t] = ldB(vbase + (size_t)(16 * t + nn) * NQ + kt + 16 * hi);
    const float* bq = bbase + (size_t)(qt + 8 * hi) * NQ + kt + nn;
    float bb0[8], bb1[8];
#pragma unroll
    for (int r = 0; r < 8; r++) {
      bb0[r] = bq[(size_t)r * NQ];
      bb1[r] = bq[(size_t)r * NQ + 16];
    }
    __builtin_prefetch(bq + 32, 0, 1);  // next bias tile toward L2
    // pipeline K fragments for next iteration
    v16bf kn[4];
    loadK((kt + 32 < NQ) ? kt + 32 : kt, kn);

    // S = Q K^T for 2 key tiles of 16
    v8f S0 = {}, S1 = {};
    S0 = wmma_bf16(qf[0], kf[0], S0);
    S0 = wmma_bf16(qf[1], kf[1], S0);
    S1 = wmma_bf16(qf[0], kf[2], S1);
    S1 = wmma_bf16(qf[1], kf[3], S1);

#pragma unroll
    for (int r = 0; r < 8; r++) { S0[r] += bb0[r]; S1[r] += bb1[r]; }

    // online softmax (fp32)
    v8f mt;
#pragma unroll
    for (int r = 0; r < 8; r++) mt[r] = fmaxf(S0[r], S1[r]);
    redmax16(mt);
    v8f alpha;
#pragma unroll
    for (int r = 0; r < 8; r++) {
      float mn = fmaxf(m[r], mt[r]);
      alpha[r] = __expf(m[r] - mn);
      m[r] = mn;
      S0[r] = __expf(S0[r] - mn);
      S1[r] = __expf(S1[r] - mn);
    }
    v8f rs;
#pragma unroll
    for (int r = 0; r < 8; r++) rs[r] = S0[r] + S1[r];
    redsum16(rs);
#pragma unroll
    for (int r = 0; r < 8; r++) l[r] = l[r] * alpha[r] + rs[r];
#pragma unroll
    for (int t = 0; t < 4; t++)
#pragma unroll
      for (int r = 0; r < 8; r++) O[t][r] *= alpha[r];

    // transpose P: C-layout -> A-layout via per-wave LDS tile
#pragma unroll
    for (int r = 0; r < 8; r++) {
      const int row = r + 8 * hi;
      myl[row * 40 + nn] = (__bf16)S0[r];
      myl[row * 40 + 16 + nn] = (__bf16)S1[r];
    }
    asm volatile("s_wait_dscnt 0" ::: "memory");
    v16bf pf = ldA(myl + nn * 40 + 16 * hi);

    // O += P V  (V^T gives contiguous B-fragments)
#pragma unroll
    for (int t = 0; t < 4; t++) O[t] = wmma_bf16(pf, vf[t], O[t]);

#pragma unroll
    for (int t = 0; t < 4; t++) kf[t] = kn[t];
  }

  // epilogue: normalize, gate, store bf16 [B,Q,H*D]
  v8f linv;
#pragma unroll
  for (int r = 0; r < 8; r++) linv[r] = 1.0f / l[r];
#pragma unroll
  for (int t = 0; t < 4; t++) {
#pragma unroll
    for (int r = 0; r < 8; r++) {
      const int q = qt + r + 8 * hi;
      const int d = 16 * t + nn;
      float val = O[t][r] * linv[r] * gbase[(size_t)q * ND + d];
      oG[((size_t)(b * NQ + q)) * NC + h * ND + d] = (__bf16)val;
    }
  }
}

// Output projection: out = oG @ w_o^T + b_o  (fp32 out)
__global__ __launch_bounds__(128) void out_proj(
    const __bf16* __restrict__ oG, const __bf16* __restrict__ woB,
    const float* __restrict__ b_o, float* __restrict__ out) {
  const int waveId = blockIdx.x * 4 + (threadIdx.x >> 5);
  const int lane = threadIdx.x & 31;
  const int nn = lane & 15, hi = lane >> 4;
  const int mt = (waveId >> 3) << 4;  // 256 M-tiles
  const int ct = (waveId & 7) << 6;   // 8 C-groups of 64
  const int kbA = 16 * hi, kbB = 16 * hi;

  const __bf16* arow = oG + (size_t)(mt + nn) * NC + kbA;
  auto loadFr = [&](int c0, v16bf& af, v16bf* bfv) {
    af = ldA(arow + c0);
#pragma unroll
    for (int t = 0; t < 4; t++)
      bfv[t] = ldB(woB + (size_t)(ct + 16 * t + nn) * NC + c0 + kbB);
  };

  v8f acc[4] = {v8f{}, v8f{}, v8f{}, v8f{}};
  v16bf afA, afB, bfA[4], bfB[4];
  loadFr(0, afA, bfA);
  for (int c0 = 0; c0 < NC; c0 += 64) {
    loadFr(c0 + 32, afB, bfB);
#pragma unroll
    for (int t = 0; t < 4; t++) acc[t] = wmma_bf16(afA, bfA[t], acc[t]);
    loadFr((c0 + 64 < NC) ? c0 + 64 : 0, afA, bfA);  // dead after last iter
#pragma unroll
    for (int t = 0; t < 4; t++) acc[t] = wmma_bf16(afB, bfB[t], acc[t]);
  }

#pragma unroll
  for (int t = 0; t < 4; t++) {
    const int c = ct + 16 * t + nn;
    const float bo = b_o[c];
#pragma unroll
    for (int r = 0; r < 8; r++) {
      const int mrow = mt + r + 8 * hi;
      out[(size_t)mrow * NC + c] = acc[t][r] + bo;
    }
  }
}

extern "C" void kernel_launch(void* const* d_in, const int* in_sizes, int n_in,
                              void* d_out, int out_size, void* d_ws, size_t ws_size,
                              hipStream_t stream) {
  const float* q_x = (const float*)d_in[0];
  // d_in[1] (kv_x) is unused by the reference path
  const float* bias = (const float*)d_in[2];
  const float* w_qkv = (const float*)d_in[3];
  const float* w_o = (const float*)d_in[4];
  const float* b_o = (const float*)d_in[5];
  const float* w_g = (const float*)d_in[6];
  const float* b_g = (const float*)d_in[7];
  float* out = (float*)d_out;
  char* ws = (char*)d_ws;

  // workspace layout (bytes)
  __bf16* qxB = (__bf16*)(ws + 0);         // 4096*512 bf16  (4 MiB)
  __bf16* WB  = (__bf16*)(ws + 4194304);   // 2048*512 bf16  (2 MiB)
  __bf16* WBg = (__bf16*)(ws + 4194304 + (size_t)1536 * 512 * 2);
  __bf16* woB = (__bf16*)(ws + 6291456);   // 512*512 bf16   (0.5 MiB)
  __bf16* qB  = (__bf16*)(ws + 6815744);   // [B,H,Q,D] bf16 (4 MiB)
  __bf16* kB  = (__bf16*)(ws + 11010048);  // [B,H,Q,D] bf16 (4 MiB)
  __bf16* vT  = (__bf16*)(ws + 15204352);  // [B,H,D,Q] bf16 (4 MiB)
  float* g    = (float*)(ws + 19398656);   // [B,H,Q,D] f32  (8 MiB)
  __bf16* oG  = (__bf16*)(ws + 27787264);  // [B,Q,H*D] bf16 (4 MiB)

  cvt_f32_bf16<<<1024, 256, 0, stream>>>(q_x, qxB, 2097152);
  cvt_f32_bf16<<<384, 256, 0, stream>>>(w_qkv, WB, 786432);
  cvt_f32_bf16<<<128, 256, 0, stream>>>(w_g, WBg, 262144);
  cvt_f32_bf16<<<128, 256, 0, stream>>>(w_o, woB, 262144);

  proj_qkvg<<<2048, 128, 0, stream>>>(qxB, WB, b_g, qB, kB, vT, g);
  attn_fused<<<512, 128, 0, stream>>>(qB, kB, vT, g, bias, oG);
  out_proj<<<512, 128, 0, stream>>>(oG, woB, b_o, out);
}
